// MTLink_33715493274244
// MI455X (gfx1250) — compile-verified, matrix operand order checked by reading, not run on hardware
//
#include <hip/hip_runtime.h>
#include <math.h>

typedef __attribute__((ext_vector_type(16))) _Float16 v16h;
typedef __attribute__((ext_vector_type(8)))  _Float16 v8h;
typedef __attribute__((ext_vector_type(8)))  float    v8f;

constexpr int Bc  = 16;
constexpr int Lc  = 512;
constexpr int Ec  = 256;
constexpr int NHc = 8;
constexpr int HDc = 32;
constexpr int TOKc = Bc * Lc;                  // 8192 tokens
constexpr float SCALEc = 0.17677669529663687f; // 1/sqrt(32)

// ---------------------------------------------------------------------------
// WMMA tile helpers (CDNA5 wave32, ISA 7.12.2 16-bit layouts).
// Per lane the 16x32 f16 fragment is two contiguous 8-half (16B) chunks:
//   K in [8*half, 8*half+7] and [16+8*half, 16+8*half+7], row = lane&15.
// ---------------------------------------------------------------------------
__device__ __forceinline__ v16h ld_tile_row(const _Float16* p, int rs) {
  const int lane = threadIdx.x & 31;
  const int half = lane >> 4;
  const int r    = lane & 15;
  const _Float16* q = p + (size_t)r * rs + 8 * half;
  const v8h lo = *(const v8h*)(q);
  const v8h hi = *(const v8h*)(q + 16);
  return __builtin_shufflevector(lo, hi, 0, 1, 2, 3, 4, 5, 6, 7,
                                 8, 9, 10, 11, 12, 13, 14, 15);
}

__device__ __forceinline__ v8f wmma_f16(v16h a, v16h b, v8f c) {
  return __builtin_amdgcn_wmma_f32_16x16x32_f16(false, a, false, b,
                                                (short)0, c, false, false);
}

// ---------------------------------------------------------------------------
// f32 -> f16 conversion (weights, once per launch)
// ---------------------------------------------------------------------------
__global__ void k_cvt_h(const float* __restrict__ src, _Float16* __restrict__ dst,
                        int n) {
  const int i = blockIdx.x * 256 + threadIdx.x;
  if (i < n) dst[i] = (_Float16)src[i];
}

// ---------------------------------------------------------------------------
// V transpose per (b,h): vt[((b*NH+h)*HD + d)*L + l] = v[(b*L+l)*v_ld + h*HD + d]
// Turns the P@V "B = V^T" fragment into contiguous b128 loads.
// ---------------------------------------------------------------------------
__global__ void k_vtrans(const _Float16* __restrict__ v, int v_ld,
                         _Float16* __restrict__ vt) {
  const int i = blockIdx.x * 256 + threadIdx.x;   // B*NH*HD*L total
  const int l = i & (Lc - 1);
  const int d = (i >> 9) & (HDc - 1);
  const int h = (i >> 14) & (NHc - 1);
  const int b = i >> 17;
  vt[i] = v[(size_t)(b * Lc + l) * v_ld + h * HDc + d];
}

// ---------------------------------------------------------------------------
// GEMM: C[M,N] = act(A[M,K] @ W[N,K]^T + bias); A,W f16, accum f32.
// One wave owns a 16x64 strip. Ping-pong software pipeline (no register
// copies): compute on fragment set 0 while set 1 loads, and vice versa.
// ---------------------------------------------------------------------------
__global__ void k_gemm_h(const _Float16* __restrict__ A, const _Float16* __restrict__ W,
                         const float* __restrict__ bias,
                         float* __restrict__ Cf, _Float16* __restrict__ Ch,
                         int M, int N, int K, int act) {
  const int wave = threadIdx.x >> 5;
  const int m0 = (blockIdx.x * 4 + wave) * 16;
  const int n0 = blockIdx.y * 64;
  if (m0 >= M) return;

  const _Float16* Abase = A + (size_t)m0 * K;

  v8f acc[4];
  {
    v8f z = {};
#pragma unroll
    for (int t = 0; t < 4; ++t) acc[t] = z;
  }

  // prologue: first K-step fragments into set 0
  v16h a0 = ld_tile_row(Abase, K);
  v16h b0[4];
#pragma unroll
  for (int t = 0; t < 4; ++t)
    b0[t] = ld_tile_row(W + (size_t)(n0 + 16 * t) * K, K);

  v16h a1;
  v16h b1[4];
  int k0 = 0;
  for (;;) {
    // ---- phase 0: load set 1 (k0+32), compute on set 0 (k0) ----
    const bool more0 = (k0 + 32 < K);
    if (more0) {
      __builtin_prefetch(Abase + k0 + 64, 0, 1);   // global_prefetch_b8
      a1 = ld_tile_row(Abase + k0 + 32, K);
#pragma unroll
      for (int t = 0; t < 4; ++t)
        b1[t] = ld_tile_row(W + (size_t)(n0 + 16 * t) * K + k0 + 32, K);
    }
#pragma unroll
    for (int t = 0; t < 4; ++t) acc[t] = wmma_f16(a0, b0[t], acc[t]);
    if (!more0) break;
    k0 += 32;

    // ---- phase 1: load set 0 (k0+32), compute on set 1 (k0) ----
    const bool more1 = (k0 + 32 < K);
    if (more1) {
      __builtin_prefetch(Abase + k0 + 64, 0, 1);
      a0 = ld_tile_row(Abase + k0 + 32, K);
#pragma unroll
      for (int t = 0; t < 4; ++t)
        b0[t] = ld_tile_row(W + (size_t)(n0 + 16 * t) * K + k0 + 32, K);
    }
#pragma unroll
    for (int t = 0; t < 4; ++t) acc[t] = wmma_f16(a1, b1[t], acc[t]);
    if (!more1) break;
    k0 += 32;
  }

  const int lane = threadIdx.x & 31, half = lane >> 4, n = lane & 15;
#pragma unroll
  for (int t = 0; t < 4; ++t) {
#pragma unroll
    for (int i = 0; i < 8; ++i) {
      const int row = m0 + i + 8 * half;
      const int col = n0 + 16 * t + n;
      float v = acc[t][i];
      if (bias) v += bias[col];
      if (act == 1) v = fmaxf(v, 0.f);
      if (Cf) Cf[(size_t)row * N + col] = v;
      if (Ch) Ch[(size_t)row * N + col] = (_Float16)v;
    }
  }
}

// ---------------------------------------------------------------------------
// Attention pass 1: per-row softmax max & sum (flash style), K-dim = HD = 32.
// Both K-tiles of a 32-wide step load before their WMMAs.
// ---------------------------------------------------------------------------
__global__ void k_attn_stats(const _Float16* __restrict__ q, int q_ld,
                             const _Float16* __restrict__ k, int k_ld,
                             const int* __restrict__ key_loc,
                             float* __restrict__ rowmax, float* __restrict__ rowsum) {
  const int bh = blockIdx.x;
  const int b = bh / NHc, h = bh % NHc;
  const int wave = threadIdx.x >> 5;
  const int q0 = (blockIdx.y * 4 + wave) * 16;
  const int lane = threadIdx.x & 31, half = lane >> 4, n = lane & 15;

  const v16h aq = ld_tile_row(q + (size_t)(b * Lc + q0) * q_ld + h * HDc, q_ld);

  float m[8], s[8];
#pragma unroll
  for (int i = 0; i < 8; ++i) { m[i] = -__builtin_inff(); s[i] = 0.f; }

  for (int k0 = 0; k0 < Lc; k0 += 32) {
    const v16h bk0 = ld_tile_row(k + (size_t)(b * Lc + k0) * k_ld + h * HDc, k_ld);
    const v16h bk1 = ld_tile_row(k + (size_t)(b * Lc + k0 + 16) * k_ld + h * HDc, k_ld);
    v8f z = {};
    const v8f c0 = wmma_f16(aq, bk0, z);
    const v8f c1 = wmma_f16(aq, bk1, z);
#pragma unroll
    for (int t = 0; t < 2; ++t) {
      const v8f& c = t ? c1 : c0;
      const int col = k0 + 16 * t + n;
      const bool kp = (key_loc[b * Lc + col] == 0);
#pragma unroll
      for (int i = 0; i < 8; ++i) {
        const float v  = kp ? -__builtin_inff() : c[i] * SCALEc;
        const float mn = fmaxf(m[i], v);
        if (mn != -__builtin_inff()) {
          s[i] = s[i] * __expf(m[i] - mn) + __expf(v - mn);
          m[i] = mn;
        }
      }
    }
  }

#pragma unroll
  for (int i = 0; i < 8; ++i) {
    float mi = m[i], si = s[i];
    for (int d = 1; d < 16; d <<= 1) {
      const float om = __shfl_xor(mi, d, 32);
      const float os = __shfl_xor(si, d, 32);
      const float mn = fmaxf(mi, om);
      if (mn != -__builtin_inff()) {
        si = si * __expf(mi - mn) + os * __expf(om - mn);
        mi = mn;
      }
    }
    if (n == 0) {
      const size_t idx = (size_t)bh * Lc + q0 + i + 8 * half;
      rowmax[idx] = mi;
      rowsum[idx] = si;
    }
  }
}

// ---------------------------------------------------------------------------
// Attention pass 2: recompute scores, exp -> f16 probs staged in LDS
// (C-tile -> A-tile relayout), accumulate P@V with WMMA using pre-transposed
// V^T (contiguous fragments). Optionally accumulate head-mean attention
// column sums via atomics (for _gen_mask).
// ---------------------------------------------------------------------------
__global__ void k_attn_pv(const _Float16* __restrict__ q, int q_ld,
                          const _Float16* __restrict__ k, int k_ld,
                          const _Float16* __restrict__ vt,  // [B*NH, HD, L]
                          const int* __restrict__ key_loc,
                          const int* __restrict__ q_loc,
                          const float* __restrict__ rowmax,
                          const float* __restrict__ rowsum,
                          _Float16* __restrict__ outh, int out_ld,
                          float* attn_key) {
  __shared__ __align__(16) _Float16 sP[4][16 * 32];
  const int bh = blockIdx.x;
  const int b = bh / NHc, h = bh % NHc;
  const int wave = threadIdx.x >> 5;
  const int q0 = (blockIdx.y * 4 + wave) * 16;
  const int lane = threadIdx.x & 31, half = lane >> 4, n = lane & 15;
  _Float16* sp = sP[wave];
  const _Float16* vbase = vt + (size_t)bh * HDc * Lc;

  const v16h aq = ld_tile_row(q + (size_t)(b * Lc + q0) * q_ld + h * HDc, q_ld);

  float mrow[8], rsin[8];
  bool qv[8];
#pragma unroll
  for (int i = 0; i < 8; ++i) {
    const int row = q0 + i + 8 * half;
    mrow[i] = rowmax[(size_t)bh * Lc + row];
    rsin[i] = 1.0f / fmaxf(rowsum[(size_t)bh * Lc + row], 1e-20f);
    qv[i]   = (q_loc[b * Lc + row] != 0);
  }

  v8f acc[2];
  { v8f z = {}; acc[0] = z; acc[1] = z; }

  for (int k0 = 0; k0 < Lc; k0 += 32) {
    const v16h bk0 = ld_tile_row(k + (size_t)(b * Lc + k0) * k_ld + h * HDc, k_ld);
    const v16h bk1 = ld_tile_row(k + (size_t)(b * Lc + k0 + 16) * k_ld + h * HDc, k_ld);
    v8f z = {};
    const v8f c0 = wmma_f16(aq, bk0, z);
    const v8f c1 = wmma_f16(aq, bk1, z);
#pragma unroll
    for (int t = 0; t < 2; ++t) {
      const v8f& c = t ? c1 : c0;
      const int col = k0 + 16 * t + n;
      const bool kp = (key_loc[b * Lc + col] == 0);
      float asum = 0.f;
#pragma unroll
      for (int i = 0; i < 8; ++i) {
        const float p = kp ? 0.f : __expf(c[i] * SCALEc - mrow[i]);
        sp[(i + 8 * half) * 32 + 16 * t + n] = (_Float16)p;
        if (attn_key && qv[i]) asum += p * rsin[i];
      }
      if (attn_key) atomicAdd(&attn_key[b * Lc + col], asum * (1.0f / NHc));
    }
    // P tile (16x32, f16) back from LDS in A layout (ds_load_b128 per chunk)
    const v16h ap  = ld_tile_row(sp, 32);
    const v16h bv0 = ld_tile_row(vbase + k0, Lc);
    const v16h bv1 = ld_tile_row(vbase + (size_t)16 * Lc + k0, Lc);
    acc[0] = wmma_f16(ap, bv0, acc[0]);
    acc[1] = wmma_f16(ap, bv1, acc[1]);
  }

#pragma unroll
  for (int t2 = 0; t2 < 2; ++t2) {
#pragma unroll
    for (int i = 0; i < 8; ++i) {
      const int row = q0 + i + 8 * half;
      outh[(size_t)(b * Lc + row) * out_ld + h * HDc + 16 * t2 + n] =
          (_Float16)(acc[t2][i] * rsin[i]);
    }
  }
}

// ---------------------------------------------------------------------------
// Embedding: tanh(cat(loc_emb, hour_emb) @ Wfc^T + b + cos(t*te_w+te_b)/16)
// ---------------------------------------------------------------------------
__global__ void k_embed(const int* __restrict__ loc, const int* __restrict__ hour,
                        const float* __restrict__ tims,
                        const float* __restrict__ loc_table,
                        const float* __restrict__ hour_table,
                        const float* __restrict__ fw, const float* __restrict__ fb,
                        const float* __restrict__ tew, const float* __restrict__ teb,
                        float* __restrict__ outf, _Float16* __restrict__ outh) {
  const int tok = blockIdx.x;
  const int e = threadIdx.x;
  __shared__ float cat[320];
  const int lc = loc[tok];
  const int hr = hour[tok];
  cat[e] = loc_table[(size_t)lc * Ec + e];
  if (e < 64) cat[Ec + e] = hour_table[hr * 64 + e];
  __syncthreads();
  float acc = fb[e];
  const float* wrow = fw + (size_t)e * 320;
  for (int j = 0; j < 320; ++j) acc += wrow[j] * cat[j];
  const float pos = cosf(tims[tok] * tew[e] + teb[e]) * 0.0625f; // sqrt(1/256)
  const float r = tanhf(acc + pos);
  outf[(size_t)tok * Ec + e] = r;
  outh[(size_t)tok * Ec + e] = (_Float16)r;
}

// ---------------------------------------------------------------------------
// LayerNorm over E=256, fused residual; emits f32 and f16 copies
// ---------------------------------------------------------------------------
__global__ void k_ln(const float* __restrict__ x, const float* __restrict__ res,
                     const float* __restrict__ g, const float* __restrict__ bta,
                     float* __restrict__ outf, _Float16* __restrict__ outh,
                     float eps) {
  const int row = blockIdx.x;
  const int e = threadIdx.x;
  __shared__ float sred[Ec];
  float v = x[(size_t)row * Ec + e];
  if (res) v += res[(size_t)row * Ec + e];
  sred[e] = v; __syncthreads();
  for (int s = Ec / 2; s > 0; s >>= 1) { if (e < s) sred[e] += sred[e + s]; __syncthreads(); }
  const float mean = sred[0] * (1.0f / Ec);
  __syncthreads();
  const float d = v - mean;
  sred[e] = d * d; __syncthreads();
  for (int s = Ec / 2; s > 0; s >>= 1) { if (e < s) sred[e] += sred[e + s]; __syncthreads(); }
  const float var = sred[0] * (1.0f / Ec);
  const float r = d * rsqrtf(var + eps) * g[e] + bta[e];
  if (outf) outf[(size_t)row * Ec + e] = r;
  if (outh) outh[(size_t)row * Ec + e] = (_Float16)r;
}

// ---------------------------------------------------------------------------
// _gen_mask: stable double-argsort rank via O(L^2) rank count per batch
// ---------------------------------------------------------------------------
__global__ void k_genmask(const float* __restrict__ attn_key,
                          const int* __restrict__ key_loc,
                          float* __restrict__ mask) {
  const int b = blockIdx.x;
  const int t = threadIdx.x;
  __shared__ float vals[Lc];
  __shared__ int scnt[256];
  int c = 0;
  for (int kk = t; kk < Lc; kk += 256) {
    const bool pad = (key_loc[b * Lc + kk] == 0);
    c += pad ? 0 : 1;
    const float a = attn_key[b * Lc + kk];
    vals[kk] = (a != 0.0f) ? a : __builtin_inff();
  }
  scnt[t] = c;
  __syncthreads();
  for (int s = 128; s > 0; s >>= 1) { if (t < s) scnt[t] += scnt[t + s]; __syncthreads(); }
  const int kv = (int)(0.3f * (float)scnt[0]);
  for (int kk = t; kk < Lc; kk += 256) {
    const float vk = vals[kk];
    int rank = 0;
    for (int j = 0; j < Lc; ++j) {
      const float vj = vals[j];
      rank += ((vj < vk) || (vj == vk && j < kk)) ? 1 : 0;
    }
    mask[b * Lc + kk] = ((rank < kv) && (vk != __builtin_inff())) ? 1.0f : 0.0f;
  }
}

__global__ void k_maskemb(const float* __restrict__ emb, const float* __restrict__ mask,
                          const float* __restrict__ memb,
                          float* __restrict__ outf, _Float16* __restrict__ outh) {
  const int tok = blockIdx.x, e = threadIdx.x;
  const float m = mask[tok];
  const float r = (1.0f - m) * emb[(size_t)tok * Ec + e] + m * memb[e];
  outf[(size_t)tok * Ec + e] = r;
  outh[(size_t)tok * Ec + e] = (_Float16)r;
}

__global__ void k_pool(const float* __restrict__ x, const int* __restrict__ vl,
                       _Float16* __restrict__ out, int out_ld, int off) {
  const int b = blockIdx.x, e = threadIdx.x;
  const int nvl = vl[b];
  float s = 0.f;
  for (int l = 0; l < nvl; ++l) s += x[(size_t)(b * Lc + l) * Ec + e];
  out[b * out_ld + off + e] = (_Float16)(s / fmaxf((float)nvl, 1.0f));
}

__global__ void k_fc2(const float* __restrict__ h1, const float* __restrict__ w2,
                      const float* __restrict__ b2, float* __restrict__ out) {
  const int s = threadIdx.x;
  if (s < Bc) {
    float acc = b2[0];
    for (int j = 0; j < 128; ++j) acc += w2[j] * h1[s * 128 + j];
    out[s] = 1.0f / (1.0f + __expf(-acc));
  }
}

// ---------------------------------------------------------------------------
// Host orchestration
// ---------------------------------------------------------------------------
extern "C" void kernel_launch(void* const* d_in, const int* in_sizes, int n_in,
                              void* d_out, int out_size, void* d_ws, size_t ws_size,
                              hipStream_t stream) {
  (void)in_sizes; (void)n_in; (void)out_size; (void)ws_size;

  const int*   a_loc  = (const int*)  d_in[0];
  const int*   a_tim  = (const int*)  d_in[1];
  const float* a_tims = (const float*)d_in[2];
  const int*   a_len  = (const int*)  d_in[3];
  const int*   b_loc  = (const int*)  d_in[4];
  const int*   b_tim  = (const int*)  d_in[5];
  const float* b_tims = (const float*)d_in[6];
  const int*   b_len  = (const int*)  d_in[7];
  const float* loc_table  = (const float*)d_in[8];
  const float* hour_table = (const float*)d_in[9];
  const float* emb_fc_w   = (const float*)d_in[10];
  const float* emb_fc_b   = (const float*)d_in[11];
  const float* te_w       = (const float*)d_in[12];
  const float* te_b       = (const float*)d_in[13];

  const float* t_in_w  = (const float*)d_in[14];
  const float* t_in_b  = (const float*)d_in[15];
  const float* t_out_w = (const float*)d_in[16];
  const float* t_out_b = (const float*)d_in[17];
  const float* t_l1g   = (const float*)d_in[18];
  const float* t_l1b   = (const float*)d_in[19];
  const float* t_l2g   = (const float*)d_in[20];
  const float* t_l2b   = (const float*)d_in[21];
  const float* t_f1w   = (const float*)d_in[22];
  const float* t_f1b   = (const float*)d_in[23];
  const float* t_f2w   = (const float*)d_in[24];
  const float* t_f2b   = (const float*)d_in[25];
  const float* t_fng   = (const float*)d_in[26];
  const float* t_fnb   = (const float*)d_in[27];

  const float* m_in_w  = (const float*)d_in[28];
  const float* m_in_b  = (const float*)d_in[29];
  const float* m_out_w = (const float*)d_in[30];
  const float* m_out_b = (const float*)d_in[31];
  const float* m_l1g   = (const float*)d_in[32];
  const float* m_l1b   = (const float*)d_in[33];
  const float* m_l2g   = (const float*)d_in[34];
  const float* m_l2b   = (const float*)d_in[35];
  const float* m_f1w   = (const float*)d_in[36];
  const float* m_f1b   = (const float*)d_in[37];
  const float* m_f2w   = (const float*)d_in[38];
  const float* m_f2b   = (const float*)d_in[39];
  const float* m_fng   = (const float*)d_in[40];
  const float* m_fnb   = (const float*)d_in[41];

  const float* c_in_w  = (const float*)d_in[42];
  const float* c_in_b  = (const float*)d_in[43];
  const float* mask_emb = (const float*)d_in[54];
  const float* clf1_w  = (const float*)d_in[55];
  const float* clf1_b  = (const float*)d_in[56];
  const float* clf2_w  = (const float*)d_in[57];
  const float* clf2_b  = (const float*)d_in[58];

  // ---- workspace (byte allocator, 16B-aligned chunks) ----
  char* wsb = (char*)d_ws;
  size_t off = 0;
  auto allocf = [&](size_t n) {
    float* p = (float*)(wsb + off);
    off += ((n * sizeof(float)) + 15) & ~(size_t)15;
    return p;
  };
  auto alloch = [&](size_t n) {
    _Float16* p = (_Float16*)(wsb + off);
    off += ((n * sizeof(_Float16)) + 15) & ~(size_t)15;
    return p;
  };

  const size_t TE = (size_t)TOKc * Ec;
  // f32 buffers
  float* embA_f = allocf(TE);
  float* embB_f = allocf(TE);
  float* xa_f   = allocf(TE);
  float* xb_f   = allocf(TE);
  float* act0_f = allocf(TE);
  float* act1_f = allocf(TE);
  float* x1b_f  = allocf(TE);
  float* obuf_f = allocf(TE);
  float* rowmax = allocf((size_t)Bc * NHc * Lc);
  float* rowsum = allocf((size_t)Bc * NHc * Lc);
  float* akeyB  = allocf(TOKc);
  float* akeyA  = allocf(TOKc);
  float* maskB  = allocf(TOKc);
  float* maskA  = allocf(TOKc);
  float* h1     = allocf((size_t)Bc * 128);
  // f16 buffers
  _Float16* embA_h = alloch(TE);
  _Float16* embB_h = alloch(TE);
  _Float16* xa_h   = alloch(TE);
  _Float16* xb_h   = alloch(TE);
  _Float16* act0_h = alloch(TE);
  _Float16* act1_h = alloch(TE);
  _Float16* x1b_h  = alloch(TE);
  _Float16* attnb_h = alloch(TE);
  _Float16* qkv_h  = alloch((size_t)TOKc * 3 * Ec);
  _Float16* ffb_h  = alloch((size_t)TOKc * 2 * Ec);
  _Float16* vt_h   = alloch((size_t)Bc * NHc * HDc * Lc);
  _Float16* xcat_h = alloch((size_t)Bc * 2 * Ec);
  // f16 weights
  _Float16* t_in_wh  = alloch((size_t)2 * 3 * Ec * Ec);
  _Float16* t_out_wh = alloch((size_t)2 * Ec * Ec);
  _Float16* t_f1wh   = alloch((size_t)2 * 512 * Ec);
  _Float16* t_f2wh   = alloch((size_t)2 * Ec * 512);
  _Float16* m_in_wh  = alloch((size_t)2 * 3 * Ec * Ec);
  _Float16* m_out_wh = alloch((size_t)2 * Ec * Ec);
  _Float16* m_f1wh   = alloch((size_t)2 * 512 * Ec);
  _Float16* m_f2wh   = alloch((size_t)2 * Ec * 512);
  _Float16* c_in_wh  = alloch((size_t)3 * Ec * Ec);
  _Float16* clf1_wh  = alloch((size_t)128 * 2 * Ec);

  auto cvt = [&](const float* s, _Float16* d, size_t n) {
    k_cvt_h<<<dim3((unsigned)((n + 255) / 256)), dim3(256), 0, stream>>>(s, d, (int)n);
  };
  cvt(t_in_w,  t_in_wh,  (size_t)2 * 3 * Ec * Ec);
  cvt(t_out_w, t_out_wh, (size_t)2 * Ec * Ec);
  cvt(t_f1w,   t_f1wh,   (size_t)2 * 512 * Ec);
  cvt(t_f2w,   t_f2wh,   (size_t)2 * Ec * 512);
  cvt(m_in_w,  m_in_wh,  (size_t)2 * 3 * Ec * Ec);
  cvt(m_out_w, m_out_wh, (size_t)2 * Ec * Ec);
  cvt(m_f1w,   m_f1wh,   (size_t)2 * 512 * Ec);
  cvt(m_f2w,   m_f2wh,   (size_t)2 * Ec * 512);
  cvt(c_in_w,  c_in_wh,  (size_t)3 * Ec * Ec);
  cvt(clf1_w,  clf1_wh,  (size_t)128 * 2 * Ec);

  auto gemm = [&](const _Float16* A, const _Float16* W, const float* bias,
                  float* Cf, _Float16* Ch, int M, int N, int K, int act) {
    dim3 g((M + 63) / 64, N / 64), blk(128);
    k_gemm_h<<<g, blk, 0, stream>>>(A, W, bias, Cf, Ch, M, N, K, act);
  };
  auto ln = [&](const float* x, const float* res, const float* g,
                const float* b, float* outf, _Float16* outh, float eps) {
    k_ln<<<dim3(TOKc), dim3(Ec), 0, stream>>>(x, res, g, b, outf, outh, eps);
  };
  auto attn = [&](const _Float16* qp, int qld, const _Float16* kp, int kld,
                  const _Float16* vp, int vld, const int* key_loc, const int* q_loc,
                  _Float16* outp, int outld, float* akey) {
    const int nvt = Bc * NHc * HDc * Lc;
    k_vtrans<<<dim3(nvt / 256), dim3(256), 0, stream>>>(vp, vld, vt_h);
    dim3 g(Bc * NHc, Lc / 64), blk(128);
    k_attn_stats<<<g, blk, 0, stream>>>(qp, qld, kp, kld, key_loc, rowmax, rowsum);
    k_attn_pv<<<g, blk, 0, stream>>>(qp, qld, kp, kld, vt_h, key_loc, q_loc,
                                     rowmax, rowsum, outp, outld, akey);
  };

  auto encoder = [&](const float* xin_f, const _Float16* xin_h,
                     float* xout_f, _Float16* xout_h, const int* loc,
                     const _Float16* in_w, const float* in_b,
                     const _Float16* out_w, const float* out_b,
                     const float* l1g, const float* l1b,
                     const float* l2g, const float* l2b,
                     const _Float16* f1w, const float* f1b,
                     const _Float16* f2w, const float* f2b,
                     const float* fng, const float* fnb,
                     int nl, int ffdim) {
    const float* cur_f = xin_f;
    const _Float16* cur_h = xin_h;
    for (int l = 0; l < nl; ++l) {
      gemm(cur_h, in_w + (size_t)l * 3 * Ec * Ec, in_b + (size_t)l * 3 * Ec,
           nullptr, qkv_h, TOKc, 3 * Ec, Ec, 0);
      attn(qkv_h, 3 * Ec, qkv_h + Ec, 3 * Ec, qkv_h + 2 * Ec, 3 * Ec, loc, loc,
           attnb_h, Ec, nullptr);
      gemm(attnb_h, out_w + (size_t)l * Ec * Ec, out_b + (size_t)l * Ec,
           obuf_f, nullptr, TOKc, Ec, Ec, 0);
      ln(cur_f, obuf_f, l1g + l * Ec, l1b + l * Ec, x1b_f, x1b_h, 1e-5f);
      gemm(x1b_h, f1w + (size_t)l * ffdim * Ec, f1b + (size_t)l * ffdim,
           nullptr, ffb_h, TOKc, ffdim, Ec, 1);
      gemm(ffb_h, f2w + (size_t)l * Ec * ffdim, f2b + (size_t)l * Ec,
           obuf_f, nullptr, TOKc, Ec, ffdim, 0);
      float* nf = (l & 1) ? act1_f : act0_f;
      _Float16* nh = (l & 1) ? act1_h : act0_h;
      ln(x1b_f, obuf_f, l2g + l * Ec, l2b + l * Ec, nf, nh, 1e-5f);
      cur_f = nf; cur_h = nh;
    }
    ln(cur_f, nullptr, fng, fnb, xout_f, xout_h, 1e-6f);
  };

  // 1) embeddings
  k_embed<<<TOKc, Ec, 0, stream>>>(a_loc, a_tim, a_tims, loc_table, hour_table,
                                   emb_fc_w, emb_fc_b, te_w, te_b, embA_f, embA_h);
  k_embed<<<TOKc, Ec, 0, stream>>>(b_loc, b_tim, b_tims, loc_table, hour_table,
                                   emb_fc_w, emb_fc_b, te_w, te_b, embB_f, embB_h);

  // 2) t_ encoders
  encoder(embA_f, embA_h, xa_f, xa_h, a_loc, t_in_wh, t_in_b, t_out_wh, t_out_b,
          t_l1g, t_l1b, t_l2g, t_l2b, t_f1wh, t_f1b, t_f2wh, t_f2b,
          t_fng, t_fnb, 2, 512);
  encoder(embB_f, embB_h, xb_f, xb_h, b_loc, t_in_wh, t_in_b, t_out_wh, t_out_b,
          t_l1g, t_l1b, t_l2g, t_l2b, t_f1wh, t_f1b, t_f2wh, t_f2b,
          t_fng, t_fnb, 2, 512);

  // 3) c_ cross-attention (only attention probs are needed for _gen_mask)
  // a queries over b keys -> column sums drive mask for b
  gemm(xa_h, c_in_wh, c_in_b, nullptr, qkv_h, TOKc, Ec, Ec, 0);          // Q (ld 256)
  gemm(xb_h, c_in_wh + (size_t)Ec * Ec, c_in_b + Ec, nullptr,
       qkv_h + (size_t)TOKc * Ec, TOKc, 2 * Ec, Ec, 0);                  // [K|V] (ld 512)
  hipMemsetAsync(akeyB, 0, TOKc * sizeof(float), stream);
  attn(qkv_h, Ec, qkv_h + (size_t)TOKc * Ec, 2 * Ec,
       qkv_h + (size_t)TOKc * Ec + Ec, 2 * Ec, b_loc, a_loc, attnb_h, Ec, akeyB);

  // b queries over a keys -> mask for a
  gemm(xb_h, c_in_wh, c_in_b, nullptr, qkv_h, TOKc, Ec, Ec, 0);
  gemm(xa_h, c_in_wh + (size_t)Ec * Ec, c_in_b + Ec, nullptr,
       qkv_h + (size_t)TOKc * Ec, TOKc, 2 * Ec, Ec, 0);
  hipMemsetAsync(akeyA, 0, TOKc * sizeof(float), stream);
  attn(qkv_h, Ec, qkv_h + (size_t)TOKc * Ec, 2 * Ec,
       qkv_h + (size_t)TOKc * Ec + Ec, 2 * Ec, a_loc, b_loc, attnb_h, Ec, akeyA);

  // 4) masks + masked embeddings (f32 in place, f16 refreshed)
  k_genmask<<<Bc, 256, 0, stream>>>(akeyB, b_loc, maskB);
  k_genmask<<<Bc, 256, 0, stream>>>(akeyA, a_loc, maskA);
  k_maskemb<<<TOKc, Ec, 0, stream>>>(embA_f, maskA, mask_emb, embA_f, embA_h);
  k_maskemb<<<TOKc, Ec, 0, stream>>>(embB_f, maskB, mask_emb, embB_f, embB_h);

  // 5) m_ encoders on masked embeddings
  encoder(embA_f, embA_h, xa_f, xa_h, a_loc, m_in_wh, m_in_b, m_out_wh, m_out_b,
          m_l1g, m_l1b, m_l2g, m_l2b, m_f1wh, m_f1b, m_f2wh, m_f2b,
          m_fng, m_fnb, 2, 512);
  encoder(embB_f, embB_h, xb_f, xb_h, b_loc, m_in_wh, m_in_b, m_out_wh, m_out_b,
          m_l1g, m_l1b, m_l2g, m_l2b, m_f1wh, m_f1b, m_f2wh, m_f2b,
          m_fng, m_fnb, 2, 512);

  // 6) pool + classifier
  k_pool<<<Bc, Ec, 0, stream>>>(xa_f, a_len, xcat_h, 2 * Ec, 0);
  k_pool<<<Bc, Ec, 0, stream>>>(xb_f, b_len, xcat_h, 2 * Ec, Ec);
  gemm(xcat_h, clf1_wh, clf1_b, h1, nullptr, Bc, 128, 2 * Ec, 1);
  k_fc2<<<1, 32, 0, stream>>>(h1, clf2_w, clf2_b, (float*)d_out);
}
